// Decomposed_GSA_65180423685509
// MI455X (gfx1250) — compile-verified, matrix operand order checked by reading, not run on hardware
//
#include <hip/hip_runtime.h>
#include <hip/hip_bf16.h>
#include <cstdint>

// ---------------------------------------------------------------------------
// Problem constants (B,H,W,C,NH) = (16,64,64,256,8), KD = 32
// ---------------------------------------------------------------------------
#define B_   16
#define H_   64
#define W_   64
#define C_   256
#define NH_  8
#define KD_  32
#define M_   (B_ * H_ * W_)          // 65536 rows for the dense GEMMs
#define SCALE_K 0.17677669529663687f // 32^-0.5

typedef __attribute__((ext_vector_type(16))) _Float16 v16h;
typedef __attribute__((ext_vector_type(8)))  float    v8f;
typedef __attribute__((ext_vector_type(4)))  float    v4f;
typedef __attribute__((ext_vector_type(4)))  unsigned int v4u;
typedef __attribute__((ext_vector_type(8)))  int      v8i;
typedef __attribute__((ext_vector_type(4)))  int      v4i;

union Frag16 { v16h v; v4f f4[2]; };

// Load a 16x32 f16 fragment in the CDNA5 WMMA A/B VGPR layout.
// Lane l: row = l&15, lane-half selects K runs [h*8, h*8+8) and [16+h*8, ...).
// Source must be row-major with K contiguous (ld in elements, 16B aligned rows).
static __device__ __forceinline__ v16h load_frag(const _Float16* base, int ld) {
  const int lane = threadIdx.x & 31;
  const int row  = lane & 15;
  const int kh   = (lane >> 4) << 3;   // 0 or 8
  const _Float16* p = base + row * ld;
  Frag16 f;
  f.f4[0] = *reinterpret_cast<const v4f*>(p + kh);
  f.f4[1] = *reinterpret_cast<const v4f*>(p + 16 + kh);
  return f.v;
}

#define WMMA_F16(a, b, c) \
  __builtin_amdgcn_wmma_f32_16x16x32_f16(false, (a), false, (b), (short)0, (c), false, false)

// ---------------------------------------------------------------------------
// fp32 -> f16 convert (weights + x staged once; halves GEMM read bandwidth)
// ---------------------------------------------------------------------------
__global__ void f32_to_f16_kernel(const float* __restrict__ src,
                                  _Float16* __restrict__ dst, int n) {
  int i = blockIdx.x * 256 + threadIdx.x;
  if (i < n) dst[i] = (_Float16)src[i];
}

// ---------------------------------------------------------------------------
// Dense GEMM: out[M x 256] = A[M x 256] @ W^T + bias via WMMA f16->f32.
// One block = 8 waves = one 32-row stripe covering all 8 column tiles.
// The shared 32x256 A stripe (16 KB) is DMA'd into LDS by the Tensor Data
// Mover (one descriptor, wave 0 issues, TENSORcnt-gated), then each wave
// reads its A fragments with ds_read_b128 while streaming B from global
// (weights are 128 KB f16 -> L2 resident).
// ---------------------------------------------------------------------------
__global__ __launch_bounds__(256) void gemm_nt_kernel(
    const _Float16* __restrict__ A,
    const _Float16* __restrict__ Wt,
    const float*    __restrict__ bias,
    float*          __restrict__ out,
    float scale) {
  __shared__ __align__(512) _Float16 lds_a[32 * C_];
  const int m0 = blockIdx.x * 32;

#if __has_builtin(__builtin_amdgcn_tensor_load_to_lds)
  // ---- TDM: stage A[m0 : m0+32, 0:256] (f16) into LDS --------------------
  if (threadIdx.x < 32) {   // wave 0 only; TDM ignores EXEC, one op per wave
    const unsigned long long ga =
        (unsigned long long)(uintptr_t)(A + (size_t)m0 * C_);
    const unsigned int ldsoff = (unsigned int)(uintptr_t)&lds_a[0];

    v4u g0;                                    // D# group 0 (ISA 8.3)
    g0[0] = 1u;                                // count=1, user mode
    g0[1] = ldsoff;                            // lds_addr (bytes)
    g0[2] = (unsigned int)ga;                  // global_addr[31:0]
    g0[3] = (unsigned int)((ga >> 32) & 0x01FFFFFFu) | (2u << 30); // addr[56:32] | type=2

    v8i g1;                                    // D# group 1 (ISA 8.4)
    g1[0] = (int)(1u << 16);                   // data_size=1 (2 bytes/elem)
    g1[1] = (int)((C_ & 0xFFFF) << 16);        // tensor_dim0[15:0] = 256
    g1[2] = (int)(((C_ >> 16) & 0xFFFF) |      // tensor_dim0[31:16]
                  ((M_ & 0xFFFF) << 16));      // tensor_dim1[15:0]
    g1[3] = (int)(((M_ >> 16) & 0xFFFF) |      // tensor_dim1[31:16] = 1
                  ((C_ & 0xFFFF) << 16));      // tile_dim0 = 256
    g1[4] = 32;                                // tile_dim1 = 32 rows
    g1[5] = C_;                                // tensor_dim0_stride = 256
    g1[6] = 0;
    g1[7] = 0;

    v4i gz = {0, 0, 0, 0};                     // groups 2/3 unused (2-D tile)
#if __clang_major__ >= 23
    v8i gz8 = {0, 0, 0, 0, 0, 0, 0, 0};
    __builtin_amdgcn_tensor_load_to_lds(g0, g1, gz, gz, gz8, 0);
#else
    __builtin_amdgcn_tensor_load_to_lds(g0, g1, gz, gz, 0);
#endif
    __builtin_amdgcn_s_wait_tensorcnt(0);
  }
#else
  // Fallback: cooperative vector copy of the A stripe into LDS.
  for (int i = threadIdx.x; i < 32 * C_ / 8; i += 256) {
    *reinterpret_cast<v4f*>(&lds_a[i * 8]) =
        *reinterpret_cast<const v4f*>(A + (size_t)m0 * C_ + i * 8);
  }
#endif
  __syncthreads();

  const int n0 = (threadIdx.x >> 5) * 32;      // wave -> 32-col tile
  v8f acc00 = {}, acc01 = {}, acc10 = {}, acc11 = {};
  const _Float16* a0p = &lds_a[0];
  const _Float16* a1p = &lds_a[16 * C_];
  const _Float16* b0p = Wt + (size_t)n0        * C_;
  const _Float16* b1p = Wt + (size_t)(n0 + 16) * C_;

#pragma unroll
  for (int kk = 0; kk < C_; kk += 32) {
    __builtin_prefetch(b0p + kk + 64, 0, 3);   // global_prefetch_b8
    v16h a0 = load_frag(a0p + kk, C_);         // ds_read_b128 x2
    v16h a1 = load_frag(a1p + kk, C_);
    v16h b0 = load_frag(b0p + kk, C_);         // global_load_b128 x2
    v16h b1 = load_frag(b1p + kk, C_);
    acc00 = WMMA_F16(a0, b0, acc00);
    acc01 = WMMA_F16(a0, b1, acc01);
    acc10 = WMMA_F16(a1, b0, acc10);
    acc11 = WMMA_F16(a1, b1, acc11);
  }

  const int lane  = threadIdx.x & 31;
  const int halfm = lane >> 4;
  const int n     = lane & 15;
#pragma unroll
  for (int r = 0; r < 8; ++r) {
    const size_t row = (size_t)(m0 + 8 * halfm + r);
    out[row        * C_ + n0 +      n] = (acc00[r] + bias[n0 + n])      * scale;
    out[row        * C_ + n0 + 16 + n] = (acc01[r] + bias[n0 + 16 + n]) * scale;
    out[(row + 16) * C_ + n0 +      n] = (acc10[r] + bias[n0 + n])      * scale;
    out[(row + 16) * C_ + n0 + 16 + n] = (acc11[r] + bias[n0 + 16 + n]) * scale;
  }
}

// ---------------------------------------------------------------------------
// RoPE + attention operand staging (one thread per channel pair).
//   qw/kw : [b][h][hd][w][kd]  f16 (W-axis attention A / B^T operands)
//   qh/kh : [b][w][hd][h][kd]  f16 (H-axis attention)
//   vwT   : [b][h][hd][kd][w]  f16 (W-attn B operand, transposed => contiguous)
// ---------------------------------------------------------------------------
__global__ void rope_prep_kernel(const float* __restrict__ q,
                                 const float* __restrict__ k,
                                 const float* __restrict__ v,
                                 const float* __restrict__ sinb,
                                 const float* __restrict__ cosb,
                                 _Float16* __restrict__ qw, _Float16* __restrict__ kw,
                                 _Float16* __restrict__ qh, _Float16* __restrict__ kh,
                                 _Float16* __restrict__ vwT) {
  const int t   = blockIdx.x * 256 + threadIdx.x;  // over B*H*W*C/2
  const int c2  = t & (C_ / 2 - 1);
  int rest      = t >> 7;
  const int w   = rest & 63; rest >>= 6;
  const int h   = rest & 63;
  const int b   = rest >> 6;
  const int c0  = c2 * 2;
  const int hd  = c0 >> 5;
  const int kd  = c0 & 31;

  const size_t src = (size_t)t * 2;   // (b,h,w,c0) flat == 2*t
  const float q0 = q[src], q1 = q[src + 1];
  const float k0 = k[src], k1 = k[src + 1];
  const float v0 = v[src], v1 = v[src + 1];

  const size_t si = ((size_t)(h * W_ + w)) * KD_ + kd;
  const float s0 = sinb[si], s1 = sinb[si + 1];
  const float cA = cosb[si], cB = cosb[si + 1];

  const float qr0 = q0 * cA - q1 * s0;
  const float qr1 = q1 * cB + q0 * s1;
  const float kr0 = k0 * cA - k1 * s0;
  const float kr1 = k1 * cB + k0 * s1;

  const size_t wi = ((((size_t)(b * H_ + h)) * NH_ + hd) * W_ + w) * KD_ + kd;
  qw[wi] = (_Float16)qr0; qw[wi + 1] = (_Float16)qr1;
  kw[wi] = (_Float16)kr0; kw[wi + 1] = (_Float16)kr1;

  const size_t hi = ((((size_t)(b * W_ + w)) * NH_ + hd) * H_ + h) * KD_ + kd;
  qh[hi] = (_Float16)qr0; qh[hi + 1] = (_Float16)qr1;
  kh[hi] = (_Float16)kr0; kh[hi + 1] = (_Float16)kr1;

  const size_t vi = ((((size_t)(b * H_ + h)) * NH_ + hd) * KD_ + kd) * W_ + w;
  vwT[vi]      = (_Float16)v0;
  vwT[vi + W_] = (_Float16)v1;
}

// ---------------------------------------------------------------------------
// Depthwise 5x5 LEPE conv (bandwidth-bound VALU kernel).
// ---------------------------------------------------------------------------
__global__ void lepe_kernel(const float* __restrict__ v,
                            const float* __restrict__ wgt,
                            const float* __restrict__ bias,
                            float* __restrict__ out) {
  const int t = blockIdx.x * 256 + threadIdx.x;   // B*H*W*C
  const int c = t & 255;
  int rest    = t >> 8;
  const int w = rest & 63; rest >>= 6;
  const int h = rest & 63;
  const int b = rest >> 6;
  float acc = bias[c];
#pragma unroll
  for (int dy = 0; dy < 5; ++dy) {
    const int y = h + dy - 2;
    if (y < 0 || y >= H_) continue;
#pragma unroll
    for (int dx = 0; dx < 5; ++dx) {
      const int x = w + dx - 2;
      if (x < 0 || x >= W_) continue;
      acc += v[(((size_t)(b * H_ + y)) * W_ + x) * C_ + c] * wgt[(dy * 5 + dx) * C_ + c];
    }
  }
  out[t] = acc;
}

// ---------------------------------------------------------------------------
// Batched 64x64 attention, one wave per (b, line, head) task.
//   S = Q K^T (16 WMMAs) -> row softmax (shfl_xor over 16-lane half-groups,
//   v_exp_f32) -> P through LDS (re-fragment as A) -> O = P V (16 WMMAs).
// mode 0: W-axis pass, O written to vhT layout [b][w][hd][kd][h]  (f16)
// mode 1: H-axis pass, O written to (B,NH,H,W,KD) flat            (f32)
// ---------------------------------------------------------------------------
#define LDP 72   // padded LDS row pitch (f16 elems); 144B keeps 16B alignment
__global__ __launch_bounds__(128) void attn64_kernel(
    const _Float16* __restrict__ Q,    // [task][64][32]
    const _Float16* __restrict__ Km,   // [task][64][32]
    const _Float16* __restrict__ VT,   // [task][32][64]
    float*          __restrict__ outF,
    _Float16*       __restrict__ outH,
    int mode) {
  __shared__ __align__(16) _Float16 lds[4][64 * LDP];
  const int wslot = threadIdx.x >> 5;
  const int task  = blockIdx.x * 4 + wslot;   // ((b*64 + outer)*8 + hd)
  const int hd    = task & 7;
  int rest        = task >> 3;
  const int outer = rest & 63;
  const int b     = rest >> 6;

  const _Float16* Qb = Q  + (size_t)task * (64 * 32);
  const _Float16* Kb = Km + (size_t)task * (64 * 32);
  const _Float16* Vb = VT + (size_t)task * (32 * 64);

  const int lane  = threadIdx.x & 31;
  const int halfm = lane >> 4;
  const int ln    = lane & 15;

  // ---- S = Q K^T ----------------------------------------------------------
  v16h aq[4], bk[4];
#pragma unroll
  for (int i = 0; i < 4; ++i) aq[i] = load_frag(Qb + i * 16 * 32, 32);
#pragma unroll
  for (int i = 0; i < 4; ++i) bk[i] = load_frag(Kb + i * 16 * 32, 32);

  v8f s[4][4];
#pragma unroll
  for (int mt = 0; mt < 4; ++mt)
#pragma unroll
    for (int nt = 0; nt < 4; ++nt) {
      v8f z = {};
      s[mt][nt] = WMMA_F16(aq[mt], bk[nt], z);
    }

  // ---- row softmax over 64 columns, P -> LDS (f16, row-major, pitch LDP) --
#pragma unroll
  for (int mt = 0; mt < 4; ++mt) {
#pragma unroll
    for (int r = 0; r < 8; ++r) {
      float mx = s[mt][0][r];
#pragma unroll
      for (int nt = 1; nt < 4; ++nt) mx = fmaxf(mx, s[mt][nt][r]);
#pragma unroll
      for (int off = 1; off < 16; off <<= 1)
        mx = fmaxf(mx, __shfl_xor(mx, off, 32));

      float e[4];
      float sum = 0.f;
#pragma unroll
      for (int nt = 0; nt < 4; ++nt) { e[nt] = __expf(s[mt][nt][r] - mx); sum += e[nt]; }
#pragma unroll
      for (int off = 1; off < 16; off <<= 1)
        sum += __shfl_xor(sum, off, 32);
      const float inv = 1.0f / sum;

      const int row = mt * 16 + 8 * halfm + r;
#pragma unroll
      for (int nt = 0; nt < 4; ++nt)
        lds[wslot][row * LDP + nt * 16 + ln] = (_Float16)(e[nt] * inv);
    }
  }
  __syncthreads();

  // ---- O = P V ------------------------------------------------------------
  v16h ap[4][2], bv[2][2];
#pragma unroll
  for (int mt = 0; mt < 4; ++mt)
#pragma unroll
    for (int kt = 0; kt < 2; ++kt)
      ap[mt][kt] = load_frag(&lds[wslot][mt * 16 * LDP + kt * 32], LDP);
#pragma unroll
  for (int n2 = 0; n2 < 2; ++n2)
#pragma unroll
    for (int kt = 0; kt < 2; ++kt)
      bv[n2][kt] = load_frag(Vb + n2 * 16 * 64 + kt * 32, 64);

  v8f o[4][2];
#pragma unroll
  for (int mt = 0; mt < 4; ++mt)
#pragma unroll
    for (int n2 = 0; n2 < 2; ++n2) {
      v8f z = {};
      z = WMMA_F16(ap[mt][0], bv[n2][0], z);
      o[mt][n2] = WMMA_F16(ap[mt][1], bv[n2][1], z);
    }

  // ---- store --------------------------------------------------------------
#pragma unroll
  for (int mt = 0; mt < 4; ++mt)
#pragma unroll
    for (int n2 = 0; n2 < 2; ++n2)
#pragma unroll
      for (int r = 0; r < 8; ++r) {
        const int m = mt * 16 + 8 * halfm + r;
        const int n = n2 * 16 + ln;
        if (mode == 0) {
          // vhT[b][w=m][hd][kd=n][h=outer]
          outH[((((size_t)(b * 64 + m)) * NH_ + hd) * KD_ + n) * H_ + outer] =
              (_Float16)o[mt][n2][r];
        } else {
          // attn(B,NH,H=m,W=outer,KD=n) flat == (B,H,W,C) flat after reshape
          outF[((((size_t)(b * NH_ + hd)) * H_ + m) * W_ + outer) * KD_ + n] =
              o[mt][n2][r];
        }
      }
}

// ---------------------------------------------------------------------------
// attn + lepe -> f16 A operand for the output projection
// ---------------------------------------------------------------------------
__global__ void add_cvt_kernel(const float* __restrict__ a,
                               const float* __restrict__ b,
                               _Float16* __restrict__ o, int n) {
  int i = blockIdx.x * 256 + threadIdx.x;
  if (i < n) o[i] = (_Float16)(a[i] + b[i]);
}

// ---------------------------------------------------------------------------
// Host-side orchestration
// ---------------------------------------------------------------------------
extern "C" void kernel_launch(void* const* d_in, const int* in_sizes, int n_in,
                              void* d_out, int out_size, void* d_ws, size_t ws_size,
                              hipStream_t stream) {
  const float* x     = (const float*)d_in[0];
  const float* sinb  = (const float*)d_in[1];
  const float* cosb  = (const float*)d_in[2];
  const float* Wq    = (const float*)d_in[3];
  const float* bq    = (const float*)d_in[4];
  const float* Wk    = (const float*)d_in[5];
  const float* bk    = (const float*)d_in[6];
  const float* Wv    = (const float*)d_in[7];
  const float* bv    = (const float*)d_in[8];
  const float* lepw  = (const float*)d_in[9];
  const float* lepb  = (const float*)d_in[10];
  const float* Wo    = (const float*)d_in[11];
  const float* bo    = (const float*)d_in[12];
  float* out = (float*)d_out;

  char* ws = (char*)d_ws;
  size_t off = 0;
  auto alloc = [&](size_t bytes) -> char* {
    char* p = ws + off;
    off += (bytes + 255) & ~(size_t)255;
    return p;
  };

  const size_t NE  = (size_t)M_ * C_;          // 16,777,216 elements
  _Float16* xh   = (_Float16*)alloc(NE * 2);
  _Float16* Wqh  = (_Float16*)alloc(C_ * C_ * 2);
  _Float16* Wkh  = (_Float16*)alloc(C_ * C_ * 2);
  _Float16* Wvh  = (_Float16*)alloc(C_ * C_ * 2);
  _Float16* Woh  = (_Float16*)alloc(C_ * C_ * 2);
  float*    qf   = (float*)alloc(NE * 4);
  float*    kf   = (float*)alloc(NE * 4);
  float*    vf   = (float*)alloc(NE * 4);
  _Float16* qw   = (_Float16*)alloc(NE * 2);
  _Float16* kw   = (_Float16*)alloc(NE * 2);
  _Float16* qh   = (_Float16*)alloc(NE * 2);
  _Float16* kh   = (_Float16*)alloc(NE * 2);
  _Float16* vwT  = (_Float16*)alloc(NE * 2);
  // Aliased scratch: these are first used only after their aliasees are dead.
  _Float16* vhT  = (_Float16*)vf;              // v (f32) dead after rope+lepe
  float*    attn = (float*)qf;                 // q (f32) dead after rope
  float*    lepe = (float*)kf;                 // k (f32) dead after rope
  _Float16* af16 = qw;                         // qw dead after W-attention

  // 1) stage f16 copies of x and the four weight matrices
  f32_to_f16_kernel<<<(int)(NE / 256), 256, 0, stream>>>(x, xh, (int)NE);
  f32_to_f16_kernel<<<C_ * C_ / 256, 256, 0, stream>>>(Wq, Wqh, C_ * C_);
  f32_to_f16_kernel<<<C_ * C_ / 256, 256, 0, stream>>>(Wk, Wkh, C_ * C_);
  f32_to_f16_kernel<<<C_ * C_ / 256, 256, 0, stream>>>(Wv, Wvh, C_ * C_);
  f32_to_f16_kernel<<<C_ * C_ / 256, 256, 0, stream>>>(Wo, Woh, C_ * C_);

  // 2) QKV projections (k pre-scaled by KD^-0.5); one block per 32-row stripe
  const int gemm_blocks = M_ / 32;             // 2048
  gemm_nt_kernel<<<gemm_blocks, 256, 0, stream>>>(xh, Wqh, bq, qf, 1.0f);
  gemm_nt_kernel<<<gemm_blocks, 256, 0, stream>>>(xh, Wkh, bk, kf, SCALE_K);
  gemm_nt_kernel<<<gemm_blocks, 256, 0, stream>>>(xh, Wvh, bv, vf, 1.0f);

  // 3) RoPE + operand staging for both attention passes
  rope_prep_kernel<<<(int)(NE / 2 / 256), 256, 0, stream>>>(
      qf, kf, vf, sinb, cosb, qw, kw, qh, kh, vwT);

  // 4) depthwise 5x5 LEPE conv (reads vf, writes over kf)
  lepe_kernel<<<(int)(NE / 256), 256, 0, stream>>>(vf, lepw, lepb, lepe);

  // 5) W-axis attention: writes vhT (overwrites vf region)
  const int attn_blocks = B_ * 64 * NH_ / 4;   // 2048
  attn64_kernel<<<attn_blocks, 128, 0, stream>>>(qw, kw, vwT, nullptr, vhT, 0);

  // 6) H-axis attention: writes attn (overwrites qf region)
  attn64_kernel<<<attn_blocks, 128, 0, stream>>>(qh, kh, vhT, attn, nullptr, 1);

  // 7) attn + lepe -> f16 (overwrites qw region)
  add_cvt_kernel<<<(int)(NE / 256), 256, 0, stream>>>(attn, lepe, af16, (int)NE);

  // 8) output projection -> d_out
  gemm_nt_kernel<<<gemm_blocks, 256, 0, stream>>>(af16, Woh, bo, out, 1.0f);

  (void)in_sizes; (void)n_in; (void)out_size; (void)ws_size;
}
